// TreeLSTM_2774548873314
// MI455X (gfx1250) — matrix operand order repaired and checked by reading, not compiled, not run on hardware
//
#include <hip/hip_runtime.h>
#include <hip/hip_bf16.h>

typedef __attribute__((ext_vector_type(16))) __bf16 v16bf;
typedef __attribute__((ext_vector_type(8)))  float  v8f;

#define NN 5824
#define HSZ 512
#define ED 352

__device__ inline float sigf(float x) { return 1.0f / (1.0f + __expf(-x)); }

// ---- WMMA operand loaders (fp32 memory -> bf16 regs); GEMM uses k = 16h + s for A and B ----
// All K used are multiples of 16; lda/ldb multiples of 4 floats -> aligned float4 loads.

__device__ inline v16bf pack16(float4 x0, float4 x1, float4 x2, float4 x3) {
  v16bf o;
  o[0]  = (__bf16)x0.x; o[1]  = (__bf16)x0.y; o[2]  = (__bf16)x0.z; o[3]  = (__bf16)x0.w;
  o[4]  = (__bf16)x1.x; o[5]  = (__bf16)x1.y; o[6]  = (__bf16)x1.z; o[7]  = (__bf16)x1.w;
  o[8]  = (__bf16)x2.x; o[9]  = (__bf16)x2.y; o[10] = (__bf16)x2.z; o[11] = (__bf16)x2.w;
  o[12] = (__bf16)x3.x; o[13] = (__bf16)x3.y; o[14] = (__bf16)x3.z; o[15] = (__bf16)x3.w;
  return o;
}
__device__ inline v16bf zero16() {
  float4 z{};
  return pack16(z, z, z, z);
}

// A operand, full 32-K slab: elements [0..7]=p[kb..kb+7], [8..15]=p[kb+16..kb+23]
__device__ inline v16bf loadA_fast(const float* __restrict__ p, int kb) {
  float4 x0 = *(const float4*)(p + kb);
  float4 x1 = *(const float4*)(p + kb + 4);
  float4 y0 = *(const float4*)(p + kb + 16);
  float4 y1 = *(const float4*)(p + kb + 20);
  __builtin_prefetch(p + kb + 32, 0, 1); // global_prefetch_b8, next K slab
  return pack16(x0, x1, y0, y1);
}
// A operand, 16-K tail: upper slab zero
__device__ inline v16bf loadA_tail(const float* __restrict__ p, int kb) {
  float4 x0 = *(const float4*)(p + kb);
  float4 x1 = *(const float4*)(p + kb + 4);
  float4 z{};
  return pack16(x0, x1, z, z);
}
// B operand ([N,K] storage), full slab: elements [0..15]=p[kb..kb+15]
__device__ inline v16bf loadB_fast(const float* __restrict__ p, int kb) {
  float4 x0 = *(const float4*)(p + kb);
  float4 x1 = *(const float4*)(p + kb + 4);
  float4 x2 = *(const float4*)(p + kb + 8);
  float4 x3 = *(const float4*)(p + kb + 12);
  return pack16(x0, x1, x2, x3);
}

// Guarded variants (edge tiles only)
__device__ inline v16bf loadA_guard(const float* __restrict__ p, int kb, bool rok, bool kfull) {
  float4 x0{}, x1{}, y0{}, y1{};
  if (rok) {
    x0 = *(const float4*)(p + kb);
    x1 = *(const float4*)(p + kb + 4);
    if (kfull) {
      y0 = *(const float4*)(p + kb + 16);
      y1 = *(const float4*)(p + kb + 20);
    }
  }
  return pack16(x0, x1, y0, y1);
}
__device__ inline v16bf loadB_guard(const float* __restrict__ p, int kb, bool ok) {
  float4 x0{}, x1{}, x2{}, x3{};
  if (ok) {
    x0 = *(const float4*)(p + kb);
    x1 = *(const float4*)(p + kb + 4);
    x2 = *(const float4*)(p + kb + 8);
    x3 = *(const float4*)(p + kb + 12);
  }
  return pack16(x0, x1, x2, x3);
}

__device__ inline void store_acc(v8f acc, int rbase, int c, float* __restrict__ C,
                                 int ldc, int M, int N, const float* __restrict__ bias,
                                 int relu) {
  if (c >= N) return;
  float bv = bias ? bias[c] : 0.0f;
#pragma unroll
  for (int j = 0; j < 8; ++j) {
    int r = rbase + j;
    if (r < M) {
      float v = acc[j] + bv;
      if (relu) v = fmaxf(v, 0.0f);
      C[(long long)r * ldc + c] = v;
    }
  }
}

// Generic batched GEMM: C[M,N] = sum_k A[M,K] * B[N,K]^T (+bias[n]) with z-batch strides.
// Requirements (all call sites satisfy): K % 16 == 0, lda/ldb % 4 == 0, 16B-aligned bases.
__global__ __launch_bounds__(32) void gemm_bf16_wmma(
    const float* __restrict__ A, const float* __restrict__ B,
    const float* __restrict__ bias, float* __restrict__ C,
    int M, int N, int K, int lda, int ldb, int ldc,
    int bz0, long long sA0, long long sA1, long long sB0, long long sB1,
    long long sC0, long long sC1, int relu)
{
  int z = blockIdx.z;
  int z0 = z % bz0, z1 = z / bz0;
  A += z0 * sA0 + z1 * sA1;
  B += z0 * sB0 + z1 * sB1;
  C += z0 * sC0 + z1 * sC1;
  int r0 = blockIdx.y << 5;
  int n0 = blockIdx.x << 5;
  int lane = threadIdx.x & 31;
  bool hi = (lane & 16) != 0;
  int l15 = lane & 15;

  int ra = r0 + l15;
  const float* pa0 = A + (long long)ra * lda;
  const float* pa1 = pa0 + (long long)16 * lda;
  int nb = n0 + l15;
  const float* pb0 = B + (long long)nb * ldb;
  const float* pb1 = pb0 + (long long)16 * ldb;

  int kbA = hi ? 16 : 0;
  int kbB = hi ? 16 : 0;

  v8f acc00 = {}, acc01 = {}, acc10 = {}, acc11 = {};

  if ((r0 + 32 <= M) && (n0 + 32 <= N)) {
    // ---- interior tile: branch-free hot loop ----
    int K32 = K & ~31;
    for (int k0 = 0; k0 < K32; k0 += 32) {
      v16bf a0 = loadB_fast(pa0, k0 + kbA);
      v16bf a1 = loadB_fast(pa1, k0 + kbA);
      v16bf b0 = loadB_fast(pb0, k0 + kbB);
      v16bf b1 = loadB_fast(pb1, k0 + kbB);
      acc00 = __builtin_amdgcn_wmma_f32_16x16x32_bf16(false, a0, false, b0, (short)0, acc00, false, false);
      acc01 = __builtin_amdgcn_wmma_f32_16x16x32_bf16(false, a0, false, b1, (short)0, acc01, false, false);
      acc10 = __builtin_amdgcn_wmma_f32_16x16x32_bf16(false, a1, false, b0, (short)0, acc10, false, false);
      acc11 = __builtin_amdgcn_wmma_f32_16x16x32_bf16(false, a1, false, b1, (short)0, acc11, false, false);
    }
    if (K & 16) {
      int k0 = K32;
      v16bf a0 = hi ? zero16() : loadB_fast(pa0, k0);
      v16bf a1 = hi ? zero16() : loadB_fast(pa1, k0);
      v16bf b0 = hi ? zero16() : loadB_fast(pb0, k0);
      v16bf b1 = hi ? zero16() : loadB_fast(pb1, k0);
      acc00 = __builtin_amdgcn_wmma_f32_16x16x32_bf16(false, a0, false, b0, (short)0, acc00, false, false);
      acc01 = __builtin_amdgcn_wmma_f32_16x16x32_bf16(false, a0, false, b1, (short)0, acc01, false, false);
      acc10 = __builtin_amdgcn_wmma_f32_16x16x32_bf16(false, a1, false, b0, (short)0, acc10, false, false);
      acc11 = __builtin_amdgcn_wmma_f32_16x16x32_bf16(false, a1, false, b1, (short)0, acc11, false, false);
    }
  } else {
    // ---- edge tile: guarded loads ----
    bool ra0ok = ra < M;
    bool ra1ok = (ra + 16) < M;
    bool nb0ok = nb < N;
    bool nb1ok = (nb + 16) < N;
    for (int k0 = 0; k0 < K; k0 += 32) {
      bool kfull = (k0 + 32 <= K);
      bool bOk = kfull || !hi; // hi-half B slab lives at k0+16..k0+31
      v16bf a0 = loadB_guard(pa0, k0 + kbA, ra0ok && bOk);
      v16bf a1 = loadB_guard(pa1, k0 + kbA, ra1ok && bOk);
      v16bf b0 = loadB_guard(pb0, k0 + kbB, nb0ok && bOk);
      v16bf b1 = loadB_guard(pb1, k0 + kbB, nb1ok && bOk);
      acc00 = __builtin_amdgcn_wmma_f32_16x16x32_bf16(false, a0, false, b0, (short)0, acc00, false, false);
      acc01 = __builtin_amdgcn_wmma_f32_16x16x32_bf16(false, a0, false, b1, (short)0, acc01, false, false);
      acc10 = __builtin_amdgcn_wmma_f32_16x16x32_bf16(false, a1, false, b0, (short)0, acc10, false, false);
      acc11 = __builtin_amdgcn_wmma_f32_16x16x32_bf16(false, a1, false, b1, (short)0, acc11, false, false);
    }
  }

  int rhi = hi ? 8 : 0;
  store_acc(acc00, r0 + rhi, n0 + l15, C, ldc, M, N, bias, relu);
  store_acc(acc01, r0 + rhi, n0 + 16 + l15, C, ldc, M, N, bias, relu);
  store_acc(acc10, r0 + 16 + rhi, n0 + l15, C, ldc, M, N, bias, relu);
  store_acc(acc11, r0 + 16 + rhi, n0 + 16 + l15, C, ldc, M, N, bias, relu);
}

// ---- elementwise / reduction kernels ----
__global__ __launch_bounds__(256) void build_x_kernel(
    const int* __restrict__ user_ids, const int* __restrict__ poi_ids,
    const int* __restrict__ cat_ids, const int* __restrict__ coo_ids,
    const int* __restrict__ time_ids, const int* __restrict__ mask,
    const float* __restrict__ user_emb, const float* __restrict__ poi_emb,
    const float* __restrict__ cat_emb, const float* __restrict__ coo_emb,
    const float* __restrict__ time_pos, float* __restrict__ x)
{
  int i = blockIdx.x * 256 + threadIdx.x;
  if (i >= NN * ED) return;
  int n = i / ED, e = i - n * ED;
  int mi = mask[n];
  float v;
  if (e < 128)      v = user_emb[user_ids[n] * mi * 128 + e];
  else if (e < 256) v = poi_emb[poi_ids[n] * mi * 128 + (e - 128)];
  else if (e < 288) v = cat_emb[cat_ids[n] * mi * 32 + (e - 256)];
  else              v = coo_emb[coo_ids[n] * mi * 64 + (e - 288)];
  v += 0.5f * time_pos[time_ids[n] * mi * ED + e];
  x[i] = v * (float)mi;
}

__global__ __launch_bounds__(256) void leaf_kernel(
    const float* __restrict__ xWiou, float* __restrict__ h, float* __restrict__ c)
{
  int i = blockIdx.x * 256 + threadIdx.x;
  if (i >= 3888 * HSZ) return;
  int n = i >> 9, col = i & 511;
  const float* row = xWiou + (long long)n * 1536;
  float iv = row[col], ov = row[512 + col], uv = row[1024 + col];
  float cl = sigf(iv) * tanhf(uv);
  c[i] = cl;
  h[i] = sigf(ov) * tanhf(cl);
}

__global__ __launch_bounds__(256) void gather_children(
    float* __restrict__ HC, const float* __restrict__ h, int m, int m3, int offn)
{
  int i = blockIdx.x * 256 + threadIdx.x;
  if (i >= m * 1536) return;
  int j = i / 1536, r = i - j * 1536;
  int k = r >> 9, col = r & 511;
  int child = offn + (j / m3) * 3 * m3 + 3 * (j % m3) + k;
  HC[i] = h[(long long)child * HSZ + col];
}

// Vt[z=(b*2+n)][d][t] = v[t,b,n,d] from QKV rows; makes P@V a [N,K]-layout GEMM.
__global__ __launch_bounds__(256) void transpose_v(
    const float* __restrict__ QKV, float* __restrict__ Vt, int m)
{
  int i = blockIdx.x * 256 + threadIdx.x;
  if (i >= 6 * 256 * m) return;
  int z = i / (256 * m);
  int r = i - z * 256 * m;
  int d = r / m, t = r - d * m;
  int b = z >> 1, nh = z & 1;
  Vt[i] = QKV[((long long)t * 3 + b) * 1536 + 1024 + nh * 256 + d];
}

__global__ __launch_bounds__(256) void softmax_rows(float* __restrict__ S, int m, float scale)
{
  __shared__ float red[256];
  long long row = blockIdx.x;
  float* p = S + row * m;
  int tid = threadIdx.x;
  float mx = -3.4e38f;
  for (int i = tid; i < m; i += 256) mx = fmaxf(mx, p[i] * scale);
  red[tid] = mx; __syncthreads();
  for (int s = 128; s > 0; s >>= 1) { if (tid < s) red[tid] = fmaxf(red[tid], red[tid + s]); __syncthreads(); }
  mx = red[0]; __syncthreads();
  float sum = 0.0f;
  for (int i = tid; i < m; i += 256) { float e = __expf(p[i] * scale - mx); p[i] = e; sum += e; }
  red[tid] = sum; __syncthreads();
  for (int s = 128; s > 0; s >>= 1) { if (tid < s) red[tid] += red[tid + s]; __syncthreads(); }
  float inv = 1.0f / red[0];
  for (int i = tid; i < m; i += 256) p[i] *= inv;
}

__global__ __launch_bounds__(256) void add_ln_512(
    const float* __restrict__ a, const float* __restrict__ b,
    const float* __restrict__ w, const float* __restrict__ bias,
    float* __restrict__ out)
{
  __shared__ float s1[256];
  __shared__ float s2[256];
  long long row = blockIdx.x;
  int tid = threadIdx.x;
  const float* pa = a + row * 512;
  const float* pb = b + row * 512;
  float v0 = pa[tid] + pb[tid];
  float v1 = pa[tid + 256] + pb[tid + 256];
  s1[tid] = v0 + v1;
  s2[tid] = v0 * v0 + v1 * v1;
  __syncthreads();
  for (int s = 128; s > 0; s >>= 1) {
    if (tid < s) { s1[tid] += s1[tid + s]; s2[tid] += s2[tid + s]; }
    __syncthreads();
  }
  float mu = s1[0] * (1.0f / 512.0f);
  float var = s2[0] * (1.0f / 512.0f) - mu * mu;
  float rstd = rsqrtf(var + 1e-5f);
  out[row * 512 + tid] = (v0 - mu) * rstd * w[tid] + bias[tid];
  out[row * 512 + tid + 256] = (v1 - mu) * rstd * w[tid + 256] + bias[tid + 256];
}

__global__ __launch_bounds__(256) void f_sigmoid(
    float* __restrict__ F, const float* __restrict__ xWf,
    const float* __restrict__ b_f, int m, int offl)
{
  int i = blockIdx.x * 256 + threadIdx.x;
  if (i >= m * 1536) return;
  int j = i / 1536, col = i - j * 1536;
  int c512 = col & 511;
  float v = F[i] + xWf[(long long)(offl + j) * HSZ + c512] + b_f[c512];
  F[i] = sigf(v);
}

__global__ __launch_bounds__(256) void combine_kernel(
    const float* __restrict__ IOU, const float* __restrict__ xWiou,
    const float* __restrict__ F, const float* cIn,
    float* h, float* c, int m, int m3, int offl, int offn)
{
  int i = blockIdx.x * 256 + threadIdx.x;
  if (i >= m * HSZ) return;
  int j = i >> 9, col = i & 511;
  int node = offl + j;
  const float* iouRow = IOU + (long long)j * 1536;
  const float* xwRow = xWiou + (long long)node * 1536;
  float iv = iouRow[col] + xwRow[col];
  float ov = iouRow[512 + col] + xwRow[512 + col];
  float uv = iouRow[1024 + col] + xwRow[1024 + col];
  int cb = offn + (j / m3) * 3 * m3 + 3 * (j % m3);
  float cr = 0.0f;
#pragma unroll
  for (int k = 0; k < 3; ++k)
    cr += F[(long long)j * 1536 + k * 512 + col] * cIn[(long long)(cb + k) * HSZ + col];
  float cn = sigf(iv) * tanhf(uv) + cr;
  h[(long long)node * HSZ + col] = sigf(ov) * tanhf(cn);
  c[(long long)node * HSZ + col] = cn;
}

// ---- host orchestration ----
static inline void launch_gemm(hipStream_t stream, const float* A, const float* B,
    const float* bias, float* C, int M, int N, int K, int lda, int ldb, int ldc,
    int batch, int bz0, long long sA0, long long sA1, long long sB0, long long sB1,
    long long sC0, long long sC1, int relu)
{
  dim3 g((N + 31) / 32, (M + 31) / 32, batch);
  gemm_bf16_wmma<<<g, dim3(32), 0, stream>>>(A, B, bias, C, M, N, K, lda, ldb, ldc,
                                             bz0, sA0, sA1, sB0, sB1, sC0, sC1, relu);
}
static inline void launch_gemm1(hipStream_t s, const float* A, const float* B,
    const float* bias, float* C, int M, int N, int K, int lda, int ldb, int ldc, int relu)
{
  launch_gemm(s, A, B, bias, C, M, N, K, lda, ldb, ldc, 1, 1, 0, 0, 0, 0, 0, 0, relu);
}

extern "C" void kernel_launch(void* const* d_in, const int* in_sizes, int n_in,
                              void* d_out, int out_size, void* d_ws, size_t ws_size,
                              hipStream_t stream) {
  (void)in_sizes; (void)n_in; (void)out_size; (void)ws_size;
  const int* user_ids = (const int*)d_in[0];
  const int* poi_ids  = (const int*)d_in[1];
  const int* cat_ids  = (const int*)d_in[2];
  const int* coo_ids  = (const int*)d_in[3];
  const int* time_ids = (const int*)d_in[4];
  const int* maskp    = (const int*)d_in[5];
  const float* user_emb = (const float*)d_in[6];
  const float* poi_emb  = (const float*)d_in[7];
  const float* cat_emb  = (const float*)d_in[8];
  const float* coo_emb  = (const float*)d_in[9];
  const float* time_pos = (const float*)d_in[10];
  const float* W_iou = (const float*)d_in[11];
  const float* U_iou = (const float*)d_in[12];
  const float* b_iou = (const float*)d_in[13];
  const float* W_f   = (const float*)d_in[14];
  const float* U_f   = (const float*)d_in[15];
  const float* b_f   = (const float*)d_in[16];
  const float* qkv_w = (const float*)d_in[17];
  const float* qkv_b = (const float*)d_in[18];
  const float* out_w = (const float*)d_in[19];
  const float* out_b = (const float*)d_in[20];
  const float* ln1_w = (const float*)d_in[21];
  const float* ln1_b = (const float*)d_in[22];
  const float* ff1_w = (const float*)d_in[23];
  const float* ff1_b = (const float*)d_in[24];
  const float* ff2_w = (const float*)d_in[25];
  const float* ff2_b = (const float*)d_in[26];
  const float* ln2_w = (const float*)d_in[27];
  const float* ln2_b = (const float*)d_in[28];
  const float* dec_poi_w = (const float*)d_in[29];
  const float* dec_poi_b = (const float*)d_in[30];
  const float* dec_cat_w = (const float*)d_in[31];
  const float* dec_cat_b = (const float*)d_in[32];
  const float* dec_coo_w = (const float*)d_in[33];
  const float* dec_coo_b = (const float*)d_in[34];

  // workspace carve-out (~216 MB of fp32); all sizes multiples of 16 floats.
  float* wsp = (float*)d_ws;
  auto alloc = [&](size_t n) { float* r = wsp; wsp += n; return r; };
  const int MAXM = 1296;
  float* xbuf  = alloc((size_t)NN * ED);
  float* xWiou = alloc((size_t)NN * 1536);
  float* xWf   = alloc((size_t)NN * HSZ);
  float* hbuf  = alloc((size_t)NN * HSZ);
  float* cbuf  = alloc((size_t)NN * HSZ);
  float* HC0   = alloc((size_t)MAXM * 1536);
  float* XA    = alloc((size_t)MAXM * 1536);
  float* XB    = alloc((size_t)MAXM * 1536);
  float* QKV   = alloc((size_t)MAXM * 3 * 1536);
  float* Sbuf  = alloc((size_t)6 * MAXM * MAXM);
  float* Vt    = alloc((size_t)6 * 256 * MAXM);
  float* ATT   = alloc((size_t)MAXM * 1536);
  float* OUTb  = alloc((size_t)MAXM * 1536);
  float* FFb   = alloc((size_t)MAXM * 1536);
  float* Fb    = alloc((size_t)MAXM * 1536);
  float* IOUb  = alloc((size_t)MAXM * 1536);

  {
    int tot = NN * ED;
    build_x_kernel<<<(tot + 255) / 256, 256, 0, stream>>>(user_ids, poi_ids, cat_ids,
        coo_ids, time_ids, maskp, user_emb, poi_emb, cat_emb, coo_emb, time_pos, xbuf);
  }
  // xWiou = x @ W_iou^T ; xWf = x @ W_f^T  (no bias)
  launch_gemm1(stream, xbuf, W_iou, nullptr, xWiou, NN, 1536, ED, ED, ED, 1536, 0);
  launch_gemm1(stream, xbuf, W_f, nullptr, xWf, NN, HSZ, ED, ED, ED, HSZ, 0);
  {
    int tot = 3888 * HSZ;
    leaf_kernel<<<(tot + 255) / 256, 256, 0, stream>>>(xWiou, hbuf, cbuf);
  }

  const int P3[6]  = {1, 3, 9, 27, 81, 243};
  const int OFF[6] = {5808, 5760, 5616, 5184, 3888, 0};
  for (int l = 4; l >= 0; --l) {
    int m = 16 * P3[l];
    int m3 = P3[l];
    int offl = OFF[l];
    int offn = OFF[l + 1];
    int tot = m * 1536;
    gather_children<<<(tot + 255) / 256, 256, 0, stream>>>(HC0, hbuf, m, m3, offn);
    // forget-gate pre-activation: h_ch[m,1536] @ U_f^T
    launch_gemm1(stream, HC0, U_f, nullptr, Fb, m, 1536, 1536, 1536, 1536, 1536, 0);
    // 2-layer post-norm transformer, seq = m, batch = 3 (node-major [m,3,512])
    float* Xcur = HC0;
    for (int layer = 0; layer < 2; ++layer) {
      const float* qw = qkv_w + (size_t)layer * 1536 * 512;
      const float* qb = qkv_b + (size_t)layer * 1536;
      launch_gemm1(stream, Xcur, qw, qb, QKV, 3 * m, 1536, 512, 512, 512, 1536, 0);
      // scores[z=b*2+n][s][t] = Q.K^T ; Q/K rows stride 3*1536, offsets n*256 + b*1536
      launch_gemm(stream, QKV, QKV + 512, nullptr, Sbuf, m, m, 256, 4608, 4608, m,
                  6, 2, 256, 1536, 256, 1536, (long long)m * m, 2LL * m * m, 0);
      softmax_rows<<<6 * m, 256, 0, stream>>>(Sbuf, m, 0.0625f);
      {
        int tv = 6 * 256 * m;
        transpose_v<<<(tv + 255) / 256, 256, 0, stream>>>(QKV, Vt, m);
      }
      // O = P @ V via Vt[z][d][t]; scatter into ATT[m,3,512]
      launch_gemm(stream, Sbuf, Vt, nullptr, ATT, m, 256, m, m, m, 1536,
                  6, 2, (long long)m * m, 2LL * m * m, 256LL * m, 512LL * m, 256, 512, 0);
      launch_gemm1(stream, ATT, out_w + (size_t)layer * 512 * 512,
                   out_b + (size_t)layer * 512, OUTb, 3 * m, 512, 512, 512, 512, 512, 0);
      add_ln_512<<<3 * m, 256, 0, stream>>>(Xcur, OUTb, ln1_w + layer * 512,
                                            ln1_b + layer * 512, XA);
      launch_gemm1(stream, XA, ff1_w + (size_t)layer * 512 * 512,
                   ff1_b + (size_t)layer * 512, FFb, 3 * m, 512, 512, 512, 512, 512, 1);
      launch_gemm1(stream, FFb, ff2_w + (size_t)layer * 512 * 512,
                   ff2_b + (size_t)layer * 512, OUTb, 3 * m, 512, 512, 512, 512, 512, 0);
      add_ln_512<<<3 * m, 256, 0, stream>>>(XA, OUTb, ln2_w + layer * 512,
                                            ln2_b + layer * 512, XB);
      Xcur = XB;
    }
    // iou = xWiou[idx] + h_att[m,1536] @ U_iou^T + b_iou (xWiou added in combine)
    launch_gemm1(stream, Xcur, U_iou, b_iou, IOUb, m, 1536, 1536, 1536, 1536, 1536, 0);
    f_sigmoid<<<(tot + 255) / 256, 256, 0, stream>>>(Fb, xWf, b_f, m, offl);
    int tot2 = m * HSZ;
    combine_kernel<<<(tot2 + 255) / 256, 256, 0, stream>>>(IOUb, xWiou, Fb, cbuf,
                                                           hbuf, cbuf, m, m3, offl, offn);
  }

  // decoders -> d_out (concatenated poi | cat | coo)
  float* out = (float*)d_out;
  launch_gemm1(stream, hbuf, dec_poi_w, dec_poi_b, out, NN, 5000, 512, 512, 512, 5000, 0);
  launch_gemm1(stream, hbuf, dec_cat_w, dec_cat_b, out + (size_t)NN * 5000,
               NN, 300, 512, 512, 512, 300, 0);
  launch_gemm1(stream, hbuf, dec_coo_w, dec_coo_b, out + (size_t)NN * 5300,
               NN, 1024, 512, 512, 512, 1024, 0);
}
